// SelfAttentionMultihead_44719199486165
// MI455X (gfx1250) — matrix operand order repaired and checked by reading, not compiled, not run on hardware
//
#include <hip/hip_runtime.h>
#include <hip/hip_bf16.h>

typedef _Float16 v16h __attribute__((ext_vector_type(16)));
typedef _Float16 v8h  __attribute__((ext_vector_type(8)));
typedef float    v8f  __attribute__((ext_vector_type(8)));

namespace {

constexpr int BATCH = 128;
constexpr int T     = 256;
constexpr int C     = 384;
constexpr int H     = 6;
constexpr int Dh    = 64;                 // head dim
constexpr int M     = BATCH * T;          // 32768 rows
constexpr int N3    = 3 * C;              // 1152
constexpr size_t QKV_ELEMS = (size_t)BATCH * H * T * Dh;  // 12,582,912

constexpr int BM = 64, BN = 128, BK = 32; // block tile; 8 waves, 32x32 each

union H16 { v16h v; v8h h[2]; };

__device__ __forceinline__ v8f wmma16(v16h a, v16h b, v8f c) {
  // D = A(16x32 f16) * B(32x16 f16) + C(16x16 f32)
  return __builtin_amdgcn_wmma_f32_16x16x32_f16(false, a, false, b,
                                                (short)0, c, false, false);
}

} // namespace

// ---------------------------------------------------------------------------
// Kernel 1: qkv = x @ W_qkv + b_qkv, written as f16 Q[B,H,T,D], K[B,H,T,D],
// V^T[B,H,D,T] (V transposed so the attention P@V B-fragment is contiguous).
// Block: 64x128 output tile, 8 waves each computing 32x32 (4 WMMAs/K-step).
// ---------------------------------------------------------------------------
__global__ __launch_bounds__(256)
void qkv_gemm_kernel(const float* __restrict__ x,
                     const float* __restrict__ Wqkv,
                     const float* __restrict__ bqkv,
                     _Float16* __restrict__ qws,
                     _Float16* __restrict__ kws,
                     _Float16* __restrict__ vws)
{
  __shared__ alignas(16) _Float16 As[BM][BK];   // x tile, row-major
  __shared__ alignas(16) _Float16 Bs[BN][BK];   // W tile, transposed (n-major)

  const int tid  = threadIdx.x;
  const int wid  = tid >> 5;
  const int lane = tid & 31;
  const int l15  = lane & 15;
  const int hi   = lane >> 4;
  const int wy   = wid >> 2;     // 0..1 : 32-row wave tile
  const int wx   = wid & 3;      // 0..3 : 32-col wave tile
  const int n0   = blockIdx.x * BN;
  const int m0   = blockIdx.y * BM;

  const int a_m = tid >> 2;            // A staging: 8 consecutive floats
  const int a_k = (tid & 3) * 8;
  const int b_k = tid >> 3;            // B staging: 16 consecutive floats
  const int b_n = (tid & 7) * 16;

  v8f acc[2][2] = {};

  for (int k0 = 0; k0 < C; k0 += BK) {
    // ---- stage A: fp32 -> f16, packed b128 LDS store ---------------------
    {
      const float* xrow = x + (size_t)(m0 + a_m) * C + k0 + a_k;
      const float4 f0 = *(const float4*)(xrow);
      const float4 f1 = *(const float4*)(xrow + 4);
      v8h av;
      av[0] = (_Float16)f0.x; av[1] = (_Float16)f0.y;
      av[2] = (_Float16)f0.z; av[3] = (_Float16)f0.w;
      av[4] = (_Float16)f1.x; av[5] = (_Float16)f1.y;
      av[6] = (_Float16)f1.z; av[7] = (_Float16)f1.w;
      *(v8h*)&As[a_m][a_k] = av;
    }
    // ---- stage B transposed: Bs[n][k] = W[k0+k][n0+n] --------------------
    {
      const float* wrow = Wqkv + (size_t)(k0 + b_k) * N3 + n0 + b_n;
      const float4 f0 = *(const float4*)(wrow);
      const float4 f1 = *(const float4*)(wrow + 4);
      const float4 f2 = *(const float4*)(wrow + 8);
      const float4 f3 = *(const float4*)(wrow + 12);
      Bs[b_n +  0][b_k] = (_Float16)f0.x;
      Bs[b_n +  1][b_k] = (_Float16)f0.y;
      Bs[b_n +  2][b_k] = (_Float16)f0.z;
      Bs[b_n +  3][b_k] = (_Float16)f0.w;
      Bs[b_n +  4][b_k] = (_Float16)f1.x;
      Bs[b_n +  5][b_k] = (_Float16)f1.y;
      Bs[b_n +  6][b_k] = (_Float16)f1.z;
      Bs[b_n +  7][b_k] = (_Float16)f1.w;
      Bs[b_n +  8][b_k] = (_Float16)f2.x;
      Bs[b_n +  9][b_k] = (_Float16)f2.y;
      Bs[b_n + 10][b_k] = (_Float16)f2.z;
      Bs[b_n + 11][b_k] = (_Float16)f2.w;
      Bs[b_n + 12][b_k] = (_Float16)f3.x;
      Bs[b_n + 13][b_k] = (_Float16)f3.y;
      Bs[b_n + 14][b_k] = (_Float16)f3.z;
      Bs[b_n + 15][b_k] = (_Float16)f3.w;
    }
    if (k0 + BK < C) {
      __builtin_prefetch(x + (size_t)(m0 + a_m) * C + k0 + BK + a_k, 0, 3);
      __builtin_prefetch(Wqkv + (size_t)(k0 + BK + b_k) * N3 + n0 + b_n, 0, 3);
    }
    __syncthreads();

    // ---- fragments + 4 WMMAs (32x32 per wave) ---------------------------
    H16 a0, a1, b0, b1;
    const int am = wy * 32 + l15;
    a0.h[0] = *(const v8h*)&As[am][hi * 8];
    a0.h[1] = *(const v8h*)&As[am][16 + hi * 8];
    a1.h[0] = *(const v8h*)&As[am + 16][hi * 8];
    a1.h[1] = *(const v8h*)&As[am + 16][16 + hi * 8];
    const int bn = wx * 32 + l15;
    b0.h[0] = *(const v8h*)&Bs[bn][hi * 8];
    b0.h[1] = *(const v8h*)&Bs[bn][16 + hi * 8];
    b1.h[0] = *(const v8h*)&Bs[bn + 16][hi * 8];
    b1.h[1] = *(const v8h*)&Bs[bn + 16][16 + hi * 8];
    acc[0][0] = wmma16(a0.v, b0.v, acc[0][0]);
    acc[0][1] = wmma16(a0.v, b1.v, acc[0][1]);
    acc[1][0] = wmma16(a1.v, b0.v, acc[1][0]);
    acc[1][1] = wmma16(a1.v, b1.v, acc[1][1]);
    __syncthreads();
  }

  // ---- epilogue: bias + scatter to Q / K / V^T ---------------------------
  for (int mi = 0; mi < 2; ++mi) {
    for (int ni = 0; ni < 2; ++ni) {
      const int c    = n0 + wx * 32 + ni * 16 + l15;  // global output column
      const float bias = bqkv[c];
      const int which = c / C;                        // 0=Q, 1=K, 2=V
      const int cc = c % C;
      const int hh = cc >> 6;
      const int d  = cc & 63;
      const int mg0 = m0 + wy * 32 + mi * 16 + hi * 8; // 8 consecutive rows
      const int bb  = mg0 >> 8;                        // batch
      const int t0  = mg0 & 255;                       // token
      const v8f a = acc[mi][ni];

      if (which == 2) {
        v8h pv;
        for (int r = 0; r < 8; ++r) pv[r] = (_Float16)(a[r] + bias);
        *(v8h*)(vws + (((size_t)bb * H + hh) * Dh + d) * T + t0) = pv;
      } else {
        _Float16* dst = (which == 0) ? qws : kws;
        for (int r = 0; r < 8; ++r)
          dst[(((size_t)bb * H + hh) * T + t0 + r) * Dh + d] =
              (_Float16)(a[r] + bias);
      }
    }
  }
}

// ---------------------------------------------------------------------------
// Kernel 2: causal flash attention. One wave per (b,h, 16-query tile).
// 32 keys per iteration: two score tiles fill a complete 16x32 P A-fragment
// (no zero padding), so each P@V WMMA covers 32 keys.
// ---------------------------------------------------------------------------
__global__ __launch_bounds__(32)
void attn_kernel(const _Float16* __restrict__ qws,
                 const _Float16* __restrict__ kws,
                 const _Float16* __restrict__ vws,
                 _Float16* __restrict__ att)
{
  __shared__ alignas(16) _Float16 pt[16][40];   // 16x32 P tile (+8 pad)

  const int lane = threadIdx.x & 31;
  const int l15  = lane & 15;
  const int hi   = lane >> 4;
  const int qt   = blockIdx.x & 15;       // query tile 0..15
  const int bh   = blockIdx.x >> 4;       // 0..767
  const int bb   = bh / H;
  const int hh   = bh % H;

  // Q A-fragments (16 queries x 64 dims = 2 frags), pre-scaled by D^-0.5
  const _Float16* qrow = qws + ((size_t)bh * T + qt * 16 + l15) * Dh;
  H16 qf0, qf1;
  qf0.h[0] = *(const v8h*)(qrow + hi * 16);
  qf0.h[1] = *(const v8h*)(qrow + hi * 16 + 8);
  qf1.h[0] = *(const v8h*)(qrow + 32 + hi * 16);
  qf1.h[1] = *(const v8h*)(qrow + 32 + hi * 16 + 8);
  qf0.v = qf0.v * (_Float16)0.125f;       // 1/sqrt(64)
  qf1.v = qf1.v * (_Float16)0.125f;

  float rm[8], rl[8];
  v8f o[4] = {};                          // O tile 16x64 fp32
  for (int r = 0; r < 8; ++r) { rm[r] = -1e30f; rl[r] = 0.f; }

  const int qrowg = qt * 16 + hi * 8;     // global query row base for this lane

  for (int jj = 0; jj <= (qt >> 1); ++jj) {
    const int j0 = jj * 2;                // key tiles j0, j0+1
    const int j1 = j0 + 1;

    // K^T B-fragments for both key tiles (contiguous in d)
    const _Float16* k0r = kws + ((size_t)bh * T + j0 * 16 + l15) * Dh;
    const _Float16* k1r = kws + ((size_t)bh * T + j1 * 16 + l15) * Dh;
    H16 k0a, k0b, k1a, k1b;
    k0a.h[0] = *(const v8h*)(k0r + hi * 16);
    k0a.h[1] = *(const v8h*)(k0r + hi * 16 + 8);
    k0b.h[0] = *(const v8h*)(k0r + 32 + hi * 16);
    k0b.h[1] = *(const v8h*)(k0r + 32 + hi * 16 + 8);
    k1a.h[0] = *(const v8h*)(k1r + hi * 16);
    k1a.h[1] = *(const v8h*)(k1r + hi * 16 + 8);
    k1b.h[0] = *(const v8h*)(k1r + 32 + hi * 16);
    k1b.h[1] = *(const v8h*)(k1r + 32 + hi * 16 + 8);

    v8f sc0 = {}, sc1 = {};
    sc0 = wmma16(qf0.v, k0a.v, sc0);
    sc0 = wmma16(qf1.v, k0b.v, sc0);
    sc1 = wmma16(qf0.v, k1a.v, sc1);
    sc1 = wmma16(qf1.v, k1b.v, sc1);

    // causal mask: key column > query row -> -inf (handles diagonal and
    // the fully-masked overshoot tile when qt is even)
    float p0[8], p1[8], tmax[8];
    for (int r = 0; r < 8; ++r) {
      const int qg = qrowg + r;
      float s0 = (j0 * 16 + l15 > qg) ? -1e30f : sc0[r];
      float s1 = (j1 * 16 + l15 > qg) ? -1e30f : sc1[r];
      p0[r] = s0;
      p1[r] = s1;
      tmax[r] = fmaxf(s0, s1);
    }
    // row max over the 16 lanes of each half-wave group
    for (int off = 1; off < 16; off <<= 1)
      for (int r = 0; r < 8; ++r)
        tmax[r] = fmaxf(tmax[r], __shfl_xor(tmax[r], off, 32));

    float alpha[8], rs[8];
    for (int r = 0; r < 8; ++r) {
      const float mn = fmaxf(rm[r], tmax[r]);
      alpha[r] = __expf(rm[r] - mn);
      rm[r] = mn;
      p0[r] = __expf(p0[r] - mn);
      p1[r] = __expf(p1[r] - mn);
      rs[r] = p0[r] + p1[r];
    }
    for (int off = 1; off < 16; off <<= 1)
      for (int r = 0; r < 8; ++r)
        rs[r] += __shfl_xor(rs[r], off, 32);
    for (int r = 0; r < 8; ++r) rl[r] = rl[r] * alpha[r] + rs[r];
    for (int nt = 0; nt < 4; ++nt)
      for (int r = 0; r < 8; ++r) o[nt][r] *= alpha[r];

    // C-layout -> A-layout transpose of P (16x32) through LDS
    for (int r = 0; r < 8; ++r) {
      pt[hi * 8 + r][l15]      = (_Float16)p0[r];
      pt[hi * 8 + r][16 + l15] = (_Float16)p1[r];
    }
    __syncthreads();

    H16 pf;
    pf.h[0] = *(const v8h*)&pt[l15][hi * 16];       // K 0..7  / 16..23
    pf.h[1] = *(const v8h*)&pt[l15][hi * 16 + 8];   // K 8..15 / 24..31

    for (int nt = 0; nt < 4; ++nt) {
      const int d  = nt * 16 + l15;
      const int t0 = jj * 32 + hi * 16;             // <= 255-15, always in range
      const _Float16* vrow = vws + ((size_t)bh * Dh + d) * T + t0;
      H16 vf;
      vf.h[0] = *(const v8h*)(vrow);
      vf.h[1] = *(const v8h*)(vrow + 8);
      o[nt] = wmma16(pf.v, vf.v, o[nt]);
    }
    __syncthreads();
  }

  // normalize and write [B,T,C] f16 for the output projection
  for (int nt = 0; nt < 4; ++nt) {
    const int d = hh * Dh + nt * 16 + l15;
    for (int r = 0; r < 8; ++r) {
      const int t = qt * 16 + hi * 8 + r;
      att[((size_t)bb * T + t) * C + d] = (_Float16)(o[nt][r] / rl[r]);
    }
  }
}

// ---------------------------------------------------------------------------
// Kernel 3: out = att @ W_proj + b_proj (A already f16; fp32 output).
// ---------------------------------------------------------------------------
__global__ __launch_bounds__(256)
void proj_gemm_kernel(const _Float16* __restrict__ att,
                      const float* __restrict__ Wp,
                      const float* __restrict__ bp,
                      float* __restrict__ out)
{
  __shared__ alignas(16) _Float16 As[BM][BK];
  __shared__ alignas(16) _Float16 Bs[BN][BK];

  const int tid  = threadIdx.x;
  const int wid  = tid >> 5;
  const int lane = tid & 31;
  const int l15  = lane & 15;
  const int hi   = lane >> 4;
  const int wy   = wid >> 2;
  const int wx   = wid & 3;
  const int n0   = blockIdx.x * BN;
  const int m0   = blockIdx.y * BM;

  const int a_m = tid >> 2;            // 8 f16 per thread
  const int a_k = (tid & 3) * 8;
  const int b_k = tid >> 3;            // 16 floats per thread
  const int b_n = (tid & 7) * 16;

  v8f acc[2][2] = {};

  for (int k0 = 0; k0 < C; k0 += BK) {
    { // A already f16: single b128 copy
      const v8h f = *(const v8h*)(att + (size_t)(m0 + a_m) * C + k0 + a_k);
      *(v8h*)&As[a_m][a_k] = f;
    }
    { // B transposed, fp32 -> f16
      const float* wrow = Wp + (size_t)(k0 + b_k) * C + n0 + b_n;
      const float4 f0 = *(const float4*)(wrow);
      const float4 f1 = *(const float4*)(wrow + 4);
      const float4 f2 = *(const float4*)(wrow + 8);
      const float4 f3 = *(const float4*)(wrow + 12);
      Bs[b_n +  0][b_k] = (_Float16)f0.x;
      Bs[b_n +  1][b_k] = (_Float16)f0.y;
      Bs[b_n +  2][b_k] = (_Float16)f0.z;
      Bs[b_n +  3][b_k] = (_Float16)f0.w;
      Bs[b_n +  4][b_k] = (_Float16)f1.x;
      Bs[b_n +  5][b_k] = (_Float16)f1.y;
      Bs[b_n +  6][b_k] = (_Float16)f1.z;
      Bs[b_n +  7][b_k] = (_Float16)f1.w;
      Bs[b_n +  8][b_k] = (_Float16)f2.x;
      Bs[b_n +  9][b_k] = (_Float16)f2.y;
      Bs[b_n + 10][b_k] = (_Float16)f2.z;
      Bs[b_n + 11][b_k] = (_Float16)f2.w;
      Bs[b_n + 12][b_k] = (_Float16)f3.x;
      Bs[b_n + 13][b_k] = (_Float16)f3.y;
      Bs[b_n + 14][b_k] = (_Float16)f3.z;
      Bs[b_n + 15][b_k] = (_Float16)f3.w;
    }
    if (k0 + BK < C) {
      __builtin_prefetch(att + (size_t)(m0 + a_m) * C + k0 + BK + a_k, 0, 3);
      __builtin_prefetch(Wp + (size_t)(k0 + BK + b_k) * C + n0 + b_n, 0, 3);
    }
    __syncthreads();

    H16 a0, a1, b0, b1;
    const int am = wy * 32 + l15;
    a0.h[0] = *(const v8h*)&As[am][hi * 8];
    a0.h[1] = *(const v8h*)&As[am][16 + hi * 8];
    a1.h[0] = *(const v8h*)&As[am + 16][hi * 8];
    a1.h[1] = *(const v8h*)&As[am + 16][16 + hi * 8];
    const int bn = wx * 32 + l15;
    b0.h[0] = *(const v8h*)&Bs[bn][hi * 8];
    b0.h[1] = *(const v8h*)&Bs[bn][16 + hi * 8];
    b1.h[0] = *(const v8h*)&Bs[bn + 16][hi * 8];
    b1.h[1] = *(const v8h*)&Bs[bn + 16][16 + hi * 8];
    acc[0][0] = wmma16(a0.v, b0.v, acc[0][0]);
    acc[0][1] = wmma16(a0.v, b1.v, acc[0][1]);
    acc[1][0] = wmma16(a1.v, b0.v, acc[1][0]);
    acc[1][1] = wmma16(a1.v, b1.v, acc[1][1]);
    __syncthreads();
  }

  for (int mi = 0; mi < 2; ++mi) {
    for (int ni = 0; ni < 2; ++ni) {
      const int n_g  = n0 + wx * 32 + ni * 16 + l15;
      const float bias = bp[n_g];
      const int mg0  = m0 + wy * 32 + mi * 16 + hi * 8;
      const v8f a = acc[mi][ni];
      for (int r = 0; r < 8; ++r)
        out[(size_t)(mg0 + r) * C + n_g] = a[r] + bias;
    }
  }
}

// ---------------------------------------------------------------------------
extern "C" void kernel_launch(void* const* d_in, const int* in_sizes, int n_in,
                              void* d_out, int out_size, void* d_ws, size_t ws_size,
                              hipStream_t stream) {
  const float* x    = (const float*)d_in[0];
  const float* Wqkv = (const float*)d_in[1];
  const float* bqkv = (const float*)d_in[2];
  const float* Wp   = (const float*)d_in[3];
  const float* bp   = (const float*)d_in[4];
  float* out = (float*)d_out;

  _Float16* qws = (_Float16*)d_ws;              // [B,H,T,D]
  _Float16* kws = qws + QKV_ELEMS;              // [B,H,T,D]
  _Float16* vws = kws + QKV_ELEMS;              // [B,H,D,T] (transposed)
  _Float16* att = vws + QKV_ELEMS;              // [B,T,C]

  qkv_gemm_kernel<<<dim3(N3 / BN, M / BM), 256, 0, stream>>>(x, Wqkv, bqkv,
                                                             qws, kws, vws);
  attn_kernel<<<dim3(BATCH * H * (T / 16)), 32, 0, stream>>>(qws, kws, vws, att);
  proj_gemm_kernel<<<dim3(C / BN, M / BM), 256, 0, stream>>>(att, Wp, bp, out);
}
